// StreamingTransformerBlock_80968723464196
// MI455X (gfx1250) — matrix operand order repaired and checked by reading, not compiled, not run on hardware
//
#include <hip/hip_runtime.h>
#include <math.h>

// ---------------------------------------------------------------------------
// StreamingTransformerBlock for MI455X (gfx1250, wave32, WMMA bf16 path).
// qkv/o-proj/MoE GEMMs and attention QK^T / PV all run on
// v_wmma_f32_16x16x32_bf16; MoE computed sparsely (top-2 experts only).
// LDS layouts make every WMMA fragment two contiguous 16B chunks per lane
// (ds_load_b128). Byte-identical tile copies use GLOBAL_LOAD_ASYNC_TO_LDS
// (builtin confirmed present; params are v4i* in AS1/AS3).
// ---------------------------------------------------------------------------

typedef __attribute__((ext_vector_type(16))) __bf16        v16bf;
typedef __attribute__((ext_vector_type(8)))  __bf16        v8bf;
typedef __attribute__((ext_vector_type(8)))  float         v8f;
typedef __attribute__((ext_vector_type(4)))  unsigned int  u32x4;
typedef __attribute__((ext_vector_type(4)))  int           v4i_;

#if defined(__has_builtin)
#if __has_builtin(__builtin_amdgcn_global_load_async_to_lds_b128)
#define HAVE_ASYNC_LDS 1
#endif
#if __has_builtin(__builtin_amdgcn_s_wait_asynccnt)
#define HAVE_WAIT_ASYNC 1
#endif
#endif
#ifndef HAVE_ASYNC_LDS
#define HAVE_ASYNC_LDS 0
#endif
#ifndef HAVE_WAIT_ASYNC
#define HAVE_WAIT_ASYNC 0
#endif

typedef __attribute__((address_space(1))) v4i_* gv4i_p;  // global int4*
typedef __attribute__((address_space(3))) v4i_* lv4i_p;  // LDS int4*

constexpr int S_  = 1024;
constexpr int D_  = 1024;
constexpr int H_  = 16;
constexpr int HD_ = 64;
constexpr int E_  = 8;
constexpr int F_  = 4096;
constexpr int TD_ = 3 * D_;

#define DEVFN __device__ __forceinline__

DEVFN unsigned short f2bf(float f) {
  union { float f; unsigned u; } x; x.f = f;
  unsigned r = x.u + 0x7FFFu + ((x.u >> 16) & 1u);  // round-to-nearest-even
  return (unsigned short)(r >> 16);
}
DEVFN v8f zero8() {
  v8f z;
#pragma unroll
  for (int i = 0; i < 8; ++i) z[i] = 0.f;
  return z;
}
DEVFN v8f wmma_bf16(v16bf a, v16bf b, v8f c) {
  return __builtin_amdgcn_wmma_f32_16x16x32_bf16(false, a, false, b, (short)0, c,
                                                 false, false);
}
// Two contiguous 16-byte chunks -> one fragment register set.
DEVFN v16bf load_pair(const unsigned short* p0, const unsigned short* p1) {
  v8bf a = *(const v8bf*)p0;
  v8bf b = *(const v8bf*)p1;
  return __builtin_shufflevector(a, b, 0, 1, 2, 3, 4, 5, 6, 7, 8, 9, 10, 11, 12,
                                 13, 14, 15);
}
// A fragment (16x32 bf16, row-major, row stride lda u16, lda*2 % 16 == 0).
// Lanes 0-15 (M) hold K 0-15, lanes 16-31 hold K 16-31 (contiguous, matches B).
DEVFN v16bf load_a_frag(const unsigned short* A, int lda, int lane) {
  int m = lane & 15, hi = lane >> 4;
  const unsigned short* p = A + m * lda + 16 * hi;
  return load_pair(p, p + 8);
}
// B fragment from transposed LDS layout [n][k] (k<32), n-stride 40 u16.
// Lane (col n = lane&15) reads K = 16*hi .. 16*hi+15 as one 32-byte run.
DEVFN v16bf load_b_fragS(const unsigned short* Bsw, int lane) {
  int n = lane & 15, hi = lane >> 4;
  const unsigned short* p = Bsw + n * 40 + hi * 16;
  return load_pair(p, p + 8);
}
// 16-u16 byte-identical copy global->LDS (async path when available).
DEVFN void copy16_g2l(const unsigned short* src, unsigned short* dst) {
#if HAVE_ASYNC_LDS
  __builtin_amdgcn_global_load_async_to_lds_b128((gv4i_p)src, (lv4i_p)dst, 0, 0);
  __builtin_amdgcn_global_load_async_to_lds_b128((gv4i_p)(src + 8),
                                                 (lv4i_p)(dst + 8), 0, 0);
#else
  *(u32x4*)dst = *(const u32x4*)src;
  *(u32x4*)(dst + 8) = *(const u32x4*)(src + 8);
#endif
}
DEVFN void wait_copy16() {
#if HAVE_ASYNC_LDS
#if HAVE_WAIT_ASYNC
  __builtin_amdgcn_s_wait_asynccnt(0);
#else
  asm volatile("s_wait_asynccnt 0" ::: "memory");
#endif
#endif
}
DEVFN float gelu_tanh(float x) {  // jax.nn.gelu default (approximate=True)
  float x3 = x * x * x;
  return 0.5f * x * (1.0f + tanhf(0.7978845608028654f * (x + 0.044715f * x3)));
}

// ---------------------------------------------------------------------------
__global__ void cvt_kernel(const float* __restrict__ src,
                           unsigned short* __restrict__ dst, long n) {
  long i = ((long)blockIdx.x * blockDim.x + threadIdx.x) * 8;
  long stride = (long)gridDim.x * blockDim.x * 8;
  for (; i < n; i += stride) {
    union { u32x4 q; unsigned short u[8]; } t;
#pragma unroll
    for (int j = 0; j < 8; ++j) t.u[j] = f2bf(src[i + j]);
    *(u32x4*)(dst + i) = t.q;
  }
}
__global__ void zero_kernel(float* __restrict__ p, long n) {
  long i = (long)blockIdx.x * blockDim.x + threadIdx.x;
  long stride = (long)gridDim.x * blockDim.x;
  for (; i < n; i += stride) p[i] = 0.f;
}

// ---------------------------------------------------------------------------
// 64x128-tile bf16 GEMM, 4 waves/block, each wave owns 32x64 (2x4 WMMA accs),
// K stepped by 32 through LDS (A row-major, B transposed for b128 frag loads).
// EPI 0: C fp32 store.        EPI 3: C bf16 store.
// EPI 1: gathered-A rows, bias+gelu -> bf16 store.
// EPI 2: bias + routing-weight atomic scatter-accumulate into moe buffer.
template <int EPI>
__global__ __launch_bounds__(128) void gemm_kernel(
    const unsigned short* __restrict__ A, long aStrideZ, int lda,
    const unsigned short* __restrict__ B, long bStrideZ, int ldb,
    int M, int N, int K,
    const int* __restrict__ gatherIdx, const int* __restrict__ cntPtr,
    const float* __restrict__ bias, int biasStrideZ,
    float* __restrict__ Cf, int ldc,
    unsigned short* __restrict__ Cb, long cbStrideZ, int ldcb,
    float* __restrict__ accOut, int ldacc, const float* __restrict__ w8) {
  __shared__ __align__(16) unsigned short sA[64 * 40];   // [row][k], pad 40
  __shared__ __align__(16) unsigned short sB[128 * 40];  // [n][k],   pad 40

  const int e = blockIdx.z;
  const unsigned short* Ae = A + (long)e * aStrideZ;
  const unsigned short* Be = B + (long)e * bStrideZ;
  const int mCount = cntPtr ? cntPtr[e] : M;
  const int mBase = blockIdx.y * 64;
  if (mBase >= mCount) return;  // sparse MoE early-exit
  const int nBase = blockIdx.x * 128;
  const int* gidx = gatherIdx ? (gatherIdx + e * S_) : nullptr;

  const int tid = threadIdx.x;
  const int w = tid >> 5, lane = tid & 31;
  const int wm = (w >> 1) * 32, wn = (w & 1) * 64;

  v8f acc[2][4];
#pragma unroll
  for (int mi = 0; mi < 2; ++mi)
#pragma unroll
    for (int nb = 0; nb < 4; ++nb) acc[mi][nb] = zero8();

  for (int k0 = 0; k0 < K; k0 += 32) {
    {  // A tile 64x32: one 16-elem contiguous chunk per thread (async/b128)
      int r = tid >> 1, c0 = (tid & 1) * 16;
      int row = mBase + r;
      int rr = row < mCount ? row : (mCount - 1);
      long srow = (EPI == 1 && gidx) ? (long)gidx[rr] : (long)rr;
      copy16_g2l(Ae + srow * lda + k0 + c0, sA + r * 40 + c0);
    }
    {  // B tile 32x128 -> [n][k] transpose; pair k-rows so stores are b32
      int r0 = (tid & 15) * 2;   // key rows r0, r0+1
      int c0 = (tid >> 4) * 16;  // 8 groups x 16 cols = 128
      const unsigned short* s0 = Be + (long)(k0 + r0) * ldb + nBase + c0;
      const unsigned short* s1 = s0 + ldb;
      union { u32x4 q[2]; unsigned short u[16]; } ta, tb;
      ta.q[0] = *(const u32x4*)s0;
      ta.q[1] = *(const u32x4*)(s0 + 8);
      tb.q[0] = *(const u32x4*)s1;
      tb.q[1] = *(const u32x4*)(s1 + 8);
#pragma unroll
      for (int j = 0; j < 16; ++j)
        *(unsigned int*)(sB + (c0 + j) * 40 + r0) =
            (unsigned int)ta.u[j] | ((unsigned int)tb.u[j] << 16);
    }
    wait_copy16();
    __syncthreads();
    v16bf a0 = load_a_frag(sA + (wm + 0) * 40, 40, lane);
    v16bf a1 = load_a_frag(sA + (wm + 16) * 40, 40, lane);
#pragma unroll
    for (int nb = 0; nb < 4; ++nb) {
      v16bf bb = load_b_fragS(sB + (wn + nb * 16) * 40, lane);
      acc[0][nb] = wmma_bf16(a0, bb, acc[0][nb]);
      acc[1][nb] = wmma_bf16(a1, bb, acc[1][nb]);
    }
    __syncthreads();
  }

  const int cn = lane & 15, hi = lane >> 4;
#pragma unroll
  for (int mi = 0; mi < 2; ++mi) {
#pragma unroll
    for (int nb = 0; nb < 4; ++nb) {
      int col = nBase + wn + nb * 16 + cn;
#pragma unroll
      for (int v = 0; v < 8; ++v) {
        int row = mBase + wm + mi * 16 + v + (hi ? 8 : 0);
        if (row >= mCount) continue;
        float val = acc[mi][nb][v];
        if constexpr (EPI == 0) {
          Cf[(long)row * ldc + col] = val;
        } else if constexpr (EPI == 3) {
          Cb[(long)row * ldcb + col] = f2bf(val);
        } else if constexpr (EPI == 1) {
          val = gelu_tanh(val + bias[e * biasStrideZ + col]);
          Cb[(long)e * cbStrideZ + (long)row * ldcb + col] = f2bf(val);
        } else {
          val += bias[e * biasStrideZ + col];
          int token = gidx[row];
          float wt = w8[token * E_ + e];
          atomicAdd(&accOut[(long)token * ldacc + col], wt * val);
        }
      }
    }
  }
}

// ---------------------------------------------------------------------------
// Flash-style causal attention over a bf16 qkv buffer. Block = (head, 64 query
// rows), 4 waves; each wave owns 16 query rows. K tile row-major (async copy),
// V tile transposed [d][key] with b32 pair stores. Softmax scale is applied to
// the fp32 score accumulators (like the reference).
__global__ __launch_bounds__(128) void attn_kernel(
    const unsigned short* __restrict__ qkv, unsigned short* __restrict__ ob) {
  __shared__ __align__(16) unsigned short sK[32 * 72];     // [key][d], pad 72
  __shared__ __align__(16) unsigned short sVT[64 * 40];    // [d][key], pad 40
  __shared__ __align__(16) unsigned short sP[4][16 * 40];  // per-wave P tile

  const int head = blockIdx.y;
  const int qBase = blockIdx.x * 64;
  const int tid = threadIdx.x, w = tid >> 5, lane = tid & 31;
  const int lm = lane & 15, hi = lane >> 4;
  const int qRow0 = qBase + w * 16;
  const float scale = 0.125f;  // 1/sqrt(64)

  // Q as two 16x32 A-fragments straight from global bf16 (b128 loads).
  v16bf qa[2];
#pragma unroll
  for (int h2 = 0; h2 < 2; ++h2)
    qa[h2] = load_a_frag(qkv + (long)qRow0 * TD_ + head * HD_ + h2 * 32, TD_, lane);

  v8f oacc[4];
#pragma unroll
  for (int i = 0; i < 4; ++i) oacc[i] = zero8();
  float mrow[8], lrow[8];
#pragma unroll
  for (int v = 0; v < 8; ++v) { mrow[v] = -1e30f; lrow[v] = 0.f; }

  for (int jBase = 0; jBase < qBase + 64; jBase += 32) {  // causal early exit
    {  // K tile: byte-identical copy (async path when available)
      int r = tid >> 2, c0 = (tid & 3) * 16;
      copy16_g2l(qkv + (long)(jBase + r) * TD_ + D_ + head * HD_ + c0,
                 sK + r * 72 + c0);
    }
    {  // V tile -> sVT[d][key]; pair key-rows so stores are b32
      int k0r = (tid & 15) * 2;  // key rows k0r, k0r+1
      int c0 = (tid >> 4) * 8;   // 8 groups x 8 dims = 64
      const unsigned short* s0 =
          qkv + (long)(jBase + k0r) * TD_ + 2 * D_ + head * HD_ + c0;
      const unsigned short* s1 = s0 + TD_;
      union { u32x4 q; unsigned short u[8]; } ta, tb;
      ta.q = *(const u32x4*)s0;
      tb.q = *(const u32x4*)s1;
#pragma unroll
      for (int j = 0; j < 8; ++j)
        *(unsigned int*)(sVT + (c0 + j) * 40 + k0r) =
            (unsigned int)ta.u[j] | ((unsigned int)tb.u[j] << 16);
    }
    wait_copy16();
    __syncthreads();

    // S = Q @ K^T : two 16x16 tiles covering 32 keys (contraction = 64 dims)
    v8f sc[2];
#pragma unroll
    for (int sub = 0; sub < 2; ++sub) {
      v8f a2 = zero8();
      int key = sub * 16 + lm;
#pragma unroll
      for (int kc = 0; kc < 2; ++kc) {
        const unsigned short* p = sK + key * 72 + kc * 32 + 16 * hi;
        v16bf bb = load_pair(p, p + 8);  // B[d][key] = sK[key][d], contiguous d
        a2 = wmma_bf16(qa[kc], bb, a2);
      }
      sc[sub] = a2;
    }

    // scale + causal mask + online softmax (row of C-frag = v + 8*hi)
#pragma unroll
    for (int v = 0; v < 8; ++v) {
      int row = qRow0 + v + (hi ? 8 : 0);
      float v0 = sc[0][v] * scale;
      float v1 = sc[1][v] * scale;
      if (jBase + lm > row)      v0 = -1e30f;
      if (jBase + 16 + lm > row) v1 = -1e30f;
      float t = fmaxf(v0, v1);
      t = fmaxf(t, __shfl_xor(t, 1, 32));
      t = fmaxf(t, __shfl_xor(t, 2, 32));
      t = fmaxf(t, __shfl_xor(t, 4, 32));
      t = fmaxf(t, __shfl_xor(t, 8, 32));
      float mnew = fmaxf(mrow[v], t);
      float corr = __expf(mrow[v] - mnew);
      float p0 = __expf(v0 - mnew);
      float p1 = __expf(v1 - mnew);
      float rs = p0 + p1;
      rs += __shfl_xor(rs, 1, 32);
      rs += __shfl_xor(rs, 2, 32);
      rs += __shfl_xor(rs, 4, 32);
      rs += __shfl_xor(rs, 8, 32);
      lrow[v] = lrow[v] * corr + rs;
      mrow[v] = mnew;
#pragma unroll
      for (int ns = 0; ns < 4; ++ns) oacc[ns][v] *= corr;
      int prow = v + (hi ? 8 : 0);
      sP[w][prow * 40 + lm]      = f2bf(p0);
      sP[w][prow * 40 + 16 + lm] = f2bf(p1);
    }

    // O += P @ V (P re-read as A-fragment from per-wave LDS region)
    v16bf pa = load_a_frag(sP[w], 40, lane);
#pragma unroll
    for (int ns = 0; ns < 4; ++ns) {
      const unsigned short* p = sVT + (ns * 16 + lm) * 40 + 16 * hi;
      v16bf vb = load_pair(p, p + 8);  // B[key][d] = sVT[d][key], contiguous k
      oacc[ns] = wmma_bf16(pa, vb, oacc[ns]);
    }
    __syncthreads();
  }

#pragma unroll
  for (int v = 0; v < 8; ++v) {
    float rl = 1.0f / lrow[v];
    long row = qRow0 + v + (hi ? 8 : 0);
#pragma unroll
    for (int ns = 0; ns < 4; ++ns)
      ob[row * D_ + head * HD_ + ns * 16 + lm] = f2bf(oacc[ns][v] * rl);
  }
}

// ---------------------------------------------------------------------------
__global__ __launch_bounds__(256) void ln_kernel(const float* __restrict__ a,
                                                 const float* __restrict__ r,
                                                 const float* __restrict__ g,
                                                 const float* __restrict__ b,
                                                 float* __restrict__ outF,
                                                 unsigned short* __restrict__ outB) {
  __shared__ float red[256];
  const int row = blockIdx.x, tid = threadIdx.x;
  float xv[4];
  float s = 0.f;
#pragma unroll
  for (int i = 0; i < 4; ++i) {
    int c = tid + i * 256;
    float t = a[(long)row * D_ + c];
    if (r) t += r[(long)row * D_ + c];
    xv[i] = t;
    s += t;
  }
  red[tid] = s;
  __syncthreads();
  for (int off = 128; off > 0; off >>= 1) {
    if (tid < off) red[tid] += red[tid + off];
    __syncthreads();
  }
  float mean = red[0] * (1.0f / D_);
  __syncthreads();
  float s2 = 0.f;
#pragma unroll
  for (int i = 0; i < 4; ++i) { float d = xv[i] - mean; s2 += d * d; }
  red[tid] = s2;
  __syncthreads();
  for (int off = 128; off > 0; off >>= 1) {
    if (tid < off) red[tid] += red[tid + off];
    __syncthreads();
  }
  float rstd = rsqrtf(red[0] * (1.0f / D_) + 1e-5f);
#pragma unroll
  for (int i = 0; i < 4; ++i) {
    int c = tid + i * 256;
    float y = (xv[i] - mean) * rstd * g[c] + b[c];
    if (outF) outF[(long)row * D_ + c] = y;
    if (outB) outB[(long)row * D_ + c] = f2bf(y);
  }
}

// Router: logits = h @ wg (N=8 -> VALU), softmax, top-2, renormalize.
__global__ __launch_bounds__(256) void router_kernel(const float* __restrict__ h,
                                                     const float* __restrict__ wg,
                                                     float* __restrict__ w8) {
  __shared__ float red[256 * E_];
  const int s = blockIdx.x, tid = threadIdx.x;
  float acc[E_];
#pragma unroll
  for (int e = 0; e < E_; ++e) acc[e] = 0.f;
  for (int d = tid; d < D_; d += 256) {
    float xv = h[(long)s * D_ + d];
#pragma unroll
    for (int e = 0; e < E_; ++e) acc[e] += xv * wg[d * E_ + e];
  }
#pragma unroll
  for (int e = 0; e < E_; ++e) red[tid * E_ + e] = acc[e];
  __syncthreads();
  for (int off = 128; off > 0; off >>= 1) {
    if (tid < off) {
#pragma unroll
      for (int e = 0; e < E_; ++e) red[tid * E_ + e] += red[(tid + off) * E_ + e];
    }
    __syncthreads();
  }
  if (tid == 0) {
    float mx = -1e30f;
#pragma unroll
    for (int e = 0; e < E_; ++e) mx = fmaxf(mx, red[e]);
    float ex[E_];
#pragma unroll
    for (int e = 0; e < E_; ++e) ex[e] = __expf(red[e] - mx);
    int i0 = 0;
#pragma unroll
    for (int e = 1; e < E_; ++e) if (ex[e] > ex[i0]) i0 = e;
    int i1 = (i0 == 0) ? 1 : 0;
#pragma unroll
    for (int e = 0; e < E_; ++e) if (e != i0 && ex[e] > ex[i1]) i1 = e;
    float t = ex[i0] + ex[i1];  // /sum(probs) cancels in the renormalization
#pragma unroll
    for (int e = 0; e < E_; ++e) w8[(long)s * E_ + e] = 0.f;
    w8[(long)s * E_ + i0] = ex[i0] / t;
    w8[(long)s * E_ + i1] = ex[i1] / t;
  }
}

// Compact ordered per-expert token lists (deterministic, trivial cost).
__global__ void build_lists(const float* __restrict__ w8, int* __restrict__ cnt,
                            int* __restrict__ idx) {
  const int e = blockIdx.x;
  if (threadIdx.x != 0) return;
  int c = 0;
  for (int s = 0; s < S_; ++s)
    if (w8[(long)s * E_ + e] > 0.f) idx[e * S_ + c++] = s;
  cnt[e] = c;
}

// ---------------------------------------------------------------------------
extern "C" void kernel_launch(void* const* d_in, const int* in_sizes, int n_in,
                              void* d_out, int out_size, void* d_ws, size_t ws_size,
                              hipStream_t stream) {
  const float* x    = (const float*)d_in[0];
  const float* wqkv = (const float*)d_in[1];
  const float* wo   = (const float*)d_in[2];
  const float* g1   = (const float*)d_in[3];
  const float* b1   = (const float*)d_in[4];
  const float* wg   = (const float*)d_in[5];
  const float* w1   = (const float*)d_in[6];
  const float* bb1  = (const float*)d_in[7];
  const float* w2   = (const float*)d_in[8];
  const float* bb2  = (const float*)d_in[9];
  const float* g2   = (const float*)d_in[10];
  const float* b2   = (const float*)d_in[11];
  float* out = (float*)d_out;
  (void)in_sizes; (void)n_in; (void)out_size; (void)ws_size;

  char* base = (char*)d_ws;
  size_t off = 0;
  auto alloc = [&](size_t bytes) -> void* {
    void* p = base + off;
    off = (off + bytes + 255) & ~(size_t)255;
    return p;
  };
  // ~225 MB total scratch
  unsigned short* xb    = (unsigned short*)alloc((size_t)S_ * D_ * 2);
  unsigned short* wqkvb = (unsigned short*)alloc((size_t)D_ * TD_ * 2);
  unsigned short* wob   = (unsigned short*)alloc((size_t)D_ * D_ * 2);
  unsigned short* w1b   = (unsigned short*)alloc((size_t)E_ * D_ * F_ * 2);
  unsigned short* w2b   = (unsigned short*)alloc((size_t)E_ * F_ * D_ * 2);
  unsigned short* qkvb  = (unsigned short*)alloc((size_t)S_ * TD_ * 2);
  unsigned short* obuf  = (unsigned short*)alloc((size_t)S_ * D_ * 2);
  float*          oproj = (float*)alloc((size_t)S_ * D_ * 4);
  float*          hbuf  = (float*)alloc((size_t)S_ * D_ * 4);
  unsigned short* hb    = (unsigned short*)alloc((size_t)S_ * D_ * 2);
  float*          w8p   = (float*)alloc((size_t)S_ * E_ * 4);
  int*            cnt   = (int*)alloc((size_t)E_ * 4);
  int*            idx   = (int*)alloc((size_t)E_ * S_ * 4);
  unsigned short* h1    = (unsigned short*)alloc((size_t)E_ * S_ * F_ * 2);
  float*          moe   = (float*)alloc((size_t)S_ * D_ * 4);

  // 1) bf16 conversions (bandwidth-trivial vs 23.3 TB/s)
  cvt_kernel<<<512, 256, 0, stream>>>(x, xb, (long)S_ * D_);
  cvt_kernel<<<1024, 256, 0, stream>>>(wqkv, wqkvb, (long)D_ * TD_);
  cvt_kernel<<<512, 256, 0, stream>>>(wo, wob, (long)D_ * D_);
  cvt_kernel<<<4096, 256, 0, stream>>>(w1, w1b, (long)E_ * D_ * F_);
  cvt_kernel<<<4096, 256, 0, stream>>>(w2, w2b, (long)E_ * F_ * D_);

  // 2) qkv = x @ w_qkv (bf16 out -> attention reads bf16 directly)
  gemm_kernel<3><<<dim3(TD_ / 128, S_ / 64, 1), 128, 0, stream>>>(
      xb, 0, D_, wqkvb, 0, TD_, S_, TD_, D_,
      nullptr, nullptr, nullptr, 0, nullptr, 0, qkvb, 0, TD_, nullptr, 0, nullptr);

  // 3) causal flash attention per (head, 64-row query block)
  attn_kernel<<<dim3(S_ / 64, H_), 128, 0, stream>>>(qkvb, obuf);

  // 4) o-proj
  gemm_kernel<0><<<dim3(D_ / 128, S_ / 64, 1), 128, 0, stream>>>(
      obuf, 0, D_, wob, 0, D_, S_, D_, D_,
      nullptr, nullptr, nullptr, 0, oproj, D_, nullptr, 0, 0, nullptr, 0, nullptr);

  // 5) h = LN(x + o)
  ln_kernel<<<S_, 256, 0, stream>>>(x, oproj, g1, b1, hbuf, hb);

  // 6) router softmax + top-2 + expert token lists
  router_kernel<<<S_, 256, 0, stream>>>(hbuf, wg, w8p);
  build_lists<<<E_, 32, 0, stream>>>(w8p, cnt, idx);

  // 7) sparse MoE: h1 = gelu(gather(h) @ w1[e] + bb1[e]);
  //    moe += w8[t,e] * (h1 @ w2[e] + bb2[e])   (2 commutative addends/elem)
  zero_kernel<<<1024, 256, 0, stream>>>(moe, (long)S_ * D_);
  gemm_kernel<1><<<dim3(F_ / 128, S_ / 64, E_), 128, 0, stream>>>(
      hb, 0, D_, w1b, (long)D_ * F_, F_, S_, F_, D_,
      idx, cnt, bb1, F_, nullptr, 0, h1, (long)S_ * F_, F_, nullptr, 0, nullptr);
  gemm_kernel<2><<<dim3(D_ / 128, S_ / 64, E_), 128, 0, stream>>>(
      h1, (long)S_ * F_, F_, w2b, (long)F_ * D_, D_, S_, D_, F_,
      idx, cnt, bb2, D_, nullptr, 0, nullptr, 0, 0, moe, D_, w8p);

  // 8) out = LN(h + moe)
  ln_kernel<<<S_, 256, 0, stream>>>(hbuf, moe, g2, b2, out, nullptr);
}
